// MorphicConvolution_61186104099587
// MI455X (gfx1250) — matrix operand order repaired and checked by reading, not compiled, not run on hardware
//
#include <hip/hip_runtime.h>
#include <hip/hip_bf16.h>

typedef __attribute__((ext_vector_type(16))) __bf16 v16bf;
typedef __attribute__((ext_vector_type(8)))  float  v8f;

#define B_   4
#define C_   64
#define H_   128
#define W_   128
#define KK_  9
#define HW_  (H_ * W_)
#define NPIX_ (B_ * H_ * W_)
#define EPS_ 1e-5f

// ---------------- kernel 0: zero BN stat accumulators ----------------
__global__ void k_zero_stats(float* __restrict__ stats) {
    int t = threadIdx.x;
    if (t < 2 * C_) stats[t] = 0.0f;
}

// ------- kernel 1: depthwise 3x3 conv + bias, partial BN stats -------
__global__ void k_dwconv(const float* __restrict__ x, const float* __restrict__ dw_w,
                         const float* __restrict__ dw_b, float* __restrict__ dw,
                         float* __restrict__ stats) {
    __shared__ float s1[128], s2[128];
    int bid = blockIdx.x;             // one block per (b, c, h) row
    int b   = bid / (C_ * H_);
    int rem = bid % (C_ * H_);
    int c   = rem / H_;
    int h   = rem % H_;
    int w   = threadIdx.x;            // 128 threads = one row
    const float* xp = x + ((size_t)(b * C_ + c) * H_) * W_;
    float acc = dw_b[c];
#pragma unroll
    for (int ky = 0; ky < 3; ++ky) {
        int yy = h - 1 + ky;
        if (yy < 0 || yy >= H_) continue;
#pragma unroll
        for (int kx = 0; kx < 3; ++kx) {
            int xx = w - 1 + kx;
            if (xx < 0 || xx >= W_) continue;
            acc += xp[yy * W_ + xx] * dw_w[c * KK_ + ky * 3 + kx];
        }
    }
    dw[((size_t)(b * C_ + c) * H_ + h) * W_ + w] = acc;
    s1[w] = acc;
    s2[w] = acc * acc;
    __syncthreads();
    for (int s = 64; s > 0; s >>= 1) {
        if (w < s) { s1[w] += s1[w + s]; s2[w] += s2[w + s]; }
        __syncthreads();
    }
    if (w == 0) {
        atomicAdd(&stats[c], s1[0]);
        atomicAdd(&stats[C_ + c], s2[0]);
    }
}

// --------------- kernel 2: BN finalize -> scale/shift ----------------
__global__ void k_bnfin(const float* __restrict__ stats, const float* __restrict__ gamma,
                        const float* __restrict__ beta, float* __restrict__ scsh) {
    int c = threadIdx.x;
    if (c >= C_) return;
    float inv_n = 1.0f / (float)NPIX_;
    float mean  = stats[c] * inv_n;
    float var   = stats[C_ + c] * inv_n - mean * mean;
    float sc    = gamma[c] * rsqrtf(var + EPS_);
    scsh[c]      = sc;
    scsh[C_ + c] = beta[c] - mean * sc;
}

// ----------------- kernel 3: x_dw = dw*scale + shift -----------------
__global__ void k_xdw(const float* __restrict__ dw, const float* __restrict__ scsh,
                      float* __restrict__ xdw) {
    int i = blockIdx.x * blockDim.x + threadIdx.x;
    if (i >= B_ * C_ * HW_) return;
    int c = (i / HW_) % C_;
    xdw[i] = dw[i] * scsh[c] + scsh[C_ + c];
}

// ------- kernel 4: pre-swizzle weights into WMMA A-fragment order ----
// A-matrix 16x32 bf16 order used here (contiguous k, B reads match): lane L -> M = L&15, group g = L>>4,
// element i (0..15) -> klocal = i + g*16.
__global__ void k_prep_frags(const float* __restrict__ deform_w, const float* __restrict__ pw_w,
                             __bf16* __restrict__ dfrag, __bf16* __restrict__ pfrag) {
    const int NDF = 4 * 18 * 32 * 16;   // 4 M-tiles x 18 K-steps
    const int NPF = 2 * 2 * 32 * 16;    // 2 M-tiles x 2 K-steps
    int idx = blockIdx.x * blockDim.x + threadIdx.x;
    if (idx < NDF) {
        int i    = idx & 15;
        int lane = (idx >> 4) & 31;
        int t    = idx >> 9;
        int kk   = t % 18;
        int mt   = t / 18;
        int g    = lane >> 4;
        int o    = mt * 16 + (lane & 15);
        int klocal = i + (g << 4);
        int r  = kk * 32 + klocal;      // reduction index r = k*64 + c
        int k  = r >> 6;
        int c  = r & 63;
        dfrag[idx] = (__bf16)deform_w[(o * C_ + c) * KK_ + k];
    } else if (idx < NDF + NPF) {
        int p    = idx - NDF;
        int i    = p & 15;
        int lane = (p >> 4) & 31;
        int t    = p >> 9;
        int kk   = t & 1;
        int mt   = t >> 1;
        int g    = lane >> 4;
        int o    = mt * 16 + (lane & 15);
        int klocal = i + (g << 4);
        int cch  = kk * 32 + klocal;
        float v  = (o < 18) ? pw_w[o * C_ + cch] : 0.0f;
        pfrag[p] = (__bf16)v;
    }
}

// ------ kernel 5: pointwise 18x64 GEMM via WMMA -> deform offsets ----
__global__ void k_pointwise(const float* __restrict__ xdw, const __bf16* __restrict__ pfrag,
                            const float* __restrict__ pw_b, float* __restrict__ offb) {
    __shared__ __attribute__((aligned(32))) __bf16 sB[16 * 72]; // [n][c], pitch 72 (144B)
    int t   = threadIdx.x;            // 64 threads = 2 waves
    int bid = blockIdx.x;
    int b   = bid >> 10;              // 1024 tiles per image (128 rows x 8)
    int rem = bid & 1023;
    int h   = rem >> 3;
    int w0  = (rem & 7) << 4;
    // Stage B (64 channels x 16 pixels) in bf16.
    int n  = t & 15;
    int cg = t >> 4;                  // 0..3, 16 channels each
    const float* src = xdw + ((size_t)(b * C_ + cg * 16) * H_ + h) * W_ + w0 + n;
#pragma unroll
    for (int j = 0; j < 16; ++j)
        sB[n * 72 + cg * 16 + j] = (__bf16)src[(size_t)j * HW_];
    __syncthreads();

    int wv   = t >> 5;                // M-tile (o = wv*16 ..)
    int lane = t & 31;
    int g    = lane >> 4;
    int pn   = lane & 15;
    v8f acc = {};
#pragma unroll
    for (int kk = 0; kk < 2; ++kk) {
        v16bf a  = *(const v16bf*)(pfrag + (size_t)((wv * 2 + kk) * 32 + lane) * 16);
        v16bf bm = *(const v16bf*)(&sB[pn * 72 + kk * 32 + g * 16]);
        acc = __builtin_amdgcn_wmma_f32_16x16x32_bf16(false, a, false, bm,
                                                      (short)0, acc, false, false);
    }
#pragma unroll
    for (int v = 0; v < 8; ++v) {
        int o = wv * 16 + v + 8 * g;  // D layout: M = v + 8*g
        if (o < 18)
            offb[((size_t)(b * 18 + o) * H_ + h) * W_ + w0 + pn] = acc[v] + pw_b[o];
    }
}

// -- kernel 6: bilinear deformable sampling + 64x576 WMMA + epilogue --
__global__ void k_deform(const float* __restrict__ x, const float* __restrict__ xdw,
                         const float* __restrict__ offb, const __bf16* __restrict__ dfrag,
                         const float* __restrict__ deform_b, const float* __restrict__ skip_scale,
                         float* __restrict__ out) {
    // Sampled tile, bf16: [n][r = k*64 + c], pitch 584 (=576+8) -> 1168B rows, 16B aligned
    __shared__ __attribute__((aligned(32))) __bf16 sB[16 * 584];
    int t   = threadIdx.x;            // 128 threads = 4 waves
    int bid = blockIdx.x;
    int b   = bid >> 10;
    int rem = bid & 1023;
    int h   = rem >> 3;
    int w0  = (rem & 7) << 4;
    int n   = t & 15;
    int cg  = t >> 4;                 // 0..7, 8 channels each
    int w   = w0 + n;
    const float* xb = x + (size_t)(b * C_) * HW_;

    // ---- gather/sample once per tile; reused by all 64 output channels ----
    for (int k = 0; k < KK_; ++k) {
        int ky = k / 3, kx = k % 3;
        float oy = offb[((size_t)(b * 18 + 2 * k    ) * H_ + h) * W_ + w];
        float ox = offb[((size_t)(b * 18 + 2 * k + 1) * H_ + h) * W_ + w];
        float py = (float)h - 1.0f + (float)ky + oy;
        float px = (float)w - 1.0f + (float)kx + ox;
        float fy = floorf(py), fx = floorf(px);
        float wy = py - fy,    wx = px - fx;
        int y0 = (int)fy, x0 = (int)fx;
        int y1 = y0 + 1,  x1 = x0 + 1;
        int vy0 = (y0 >= 0) & (y0 < H_);
        int vy1 = (y1 >= 0) & (y1 < H_);
        int vx0 = (x0 >= 0) & (x0 < W_);
        int vx1 = (x1 >= 0) & (x1 < W_);
        int yc0 = min(max(y0, 0), H_ - 1), yc1 = min(max(y1, 0), H_ - 1);
        int xc0 = min(max(x0, 0), W_ - 1), xc1 = min(max(x1, 0), W_ - 1);
        float w00 = (1.0f - wy) * (1.0f - wx) * (float)(vy0 & vx0);
        float w01 = (1.0f - wy) * wx          * (float)(vy0 & vx1);
        float w10 = wy * (1.0f - wx)          * (float)(vy1 & vx0);
        float w11 = wy * wx                   * (float)(vy1 & vx1);
        int i00 = yc0 * W_ + xc0, i01 = yc0 * W_ + xc1;
        int i10 = yc1 * W_ + xc0, i11 = yc1 * W_ + xc1;
#pragma unroll
        for (int j = 0; j < 8; ++j) {
            int c = cg * 8 + j;
            const float* xch = xb + (size_t)c * HW_;
            float v = w00 * xch[i00] + w01 * xch[i01] + w10 * xch[i10] + w11 * xch[i11];
            sB[n * 584 + k * 64 + c] = (__bf16)v;
        }
    }
    __syncthreads();

    // ---- 64x16 output tile: 4 waves x (16 rows), K = 576 in 18 steps ----
    int mt   = t >> 5;                // wave id = M tile
    int lane = t & 31;
    int g    = lane >> 4;
    int pn   = lane & 15;
    v8f acc = {};
#pragma unroll
    for (int kk = 0; kk < 18; ++kk) {
        v16bf a  = *(const v16bf*)(dfrag + (size_t)((mt * 18 + kk) * 32 + lane) * 16);
        v16bf bm = *(const v16bf*)(&sB[pn * 584 + kk * 32 + g * 16]);
        acc = __builtin_amdgcn_wmma_f32_16x16x32_bf16(false, a, false, bm,
                                                      (short)0, acc, false, false);
    }
    float ss = skip_scale[0];
#pragma unroll
    for (int v = 0; v < 8; ++v) {
        int o = mt * 16 + v + 8 * g;
        size_t idx = ((size_t)(b * C_ + o) * H_ + h) * W_ + w0 + pn;
        out[idx] = acc[v] + deform_b[o] + xdw[idx] + x[idx] * ss;
    }
}

extern "C" void kernel_launch(void* const* d_in, const int* in_sizes, int n_in,
                              void* d_out, int out_size, void* d_ws, size_t ws_size,
                              hipStream_t stream) {
    (void)in_sizes; (void)n_in; (void)out_size; (void)ws_size;
    const float* x        = (const float*)d_in[0];
    const float* dw_w     = (const float*)d_in[1];
    const float* dw_b     = (const float*)d_in[2];
    const float* bn_gamma = (const float*)d_in[3];
    const float* bn_beta  = (const float*)d_in[4];
    const float* pw_w     = (const float*)d_in[5];
    const float* pw_b     = (const float*)d_in[6];
    const float* skip     = (const float*)d_in[7];
    const float* deform_w = (const float*)d_in[8];
    const float* deform_b = (const float*)d_in[9];
    float* out = (float*)d_out;

    char* ws = (char*)d_ws;
    float* stats = (float*)ws;  ws += (size_t)2 * C_ * sizeof(float);
    float* scsh  = (float*)ws;  ws += (size_t)2 * C_ * sizeof(float);
    float* dw    = (float*)ws;  ws += (size_t)B_ * C_ * HW_ * sizeof(float);
    float* xdw   = (float*)ws;  ws += (size_t)B_ * C_ * HW_ * sizeof(float);
    float* offb  = (float*)ws;  ws += (size_t)B_ * 18 * HW_ * sizeof(float);
    __bf16* dfrag = (__bf16*)ws; ws += (size_t)4 * 18 * 32 * 16 * sizeof(__bf16);
    __bf16* pfrag = (__bf16*)ws;

    k_zero_stats<<<1, 128, 0, stream>>>(stats);
    k_dwconv<<<B_ * C_ * H_, 128, 0, stream>>>(x, dw_w, dw_b, dw, stats);
    k_bnfin<<<1, 64, 0, stream>>>(stats, bn_gamma, bn_beta, scsh);
    int nel = B_ * C_ * HW_;
    k_xdw<<<(nel + 255) / 256, 256, 0, stream>>>(dw, scsh, xdw);
    int nfr = 4 * 18 * 32 * 16 + 2 * 2 * 32 * 16;
    k_prep_frags<<<(nfr + 255) / 256, 256, 0, stream>>>(deform_w, pw_w, dfrag, pfrag);
    int ntiles = B_ * H_ * (W_ / 16);   // 4096 tiles of 16 pixels
    k_pointwise<<<ntiles, 64, 0, stream>>>(xdw, pfrag, pw_b, offb);
    k_deform<<<ntiles, 128, 0, stream>>>(x, xdw, offb, dfrag, deform_b, skip, out);
}
